// Attention_60662118089205
// MI455X (gfx1250) — matrix operand order repaired and checked by reading, not compiled, not run on hardware
//
#include <hip/hip_runtime.h>

// ---------------------------------------------------------------------------
// Multi-head attention forward for MI455X (gfx1250), wave32 + WMMA f16.
// Stages:
//   1) qkv_h  = f16( x @ w_qkv )                     [8192 x 2304]
//   2) attn_h = f16( softmax(QK^T/8) V ) per head    [8192 x 768]
//   3) out    = attn_h @ w_out + b_out  (f32)        [8192 x 768]
// gfx1250 paths: v_wmma_f32_16x16x32_f16, global_load_async_to_lds_b128
// (ASYNCcnt), packed f16 converts, split wait counters.
// ---------------------------------------------------------------------------

#define BATCH   8
#define SEQ     1024
#define NHEADS  12
#define HDIM    64
#define EMB     768
#define QKVD    2304
#define ROWS    (BATCH * SEQ)          // 8192
#define L2E     1.4426950408889634f

typedef __attribute__((ext_vector_type(16))) _Float16 v16h;
typedef __attribute__((ext_vector_type(8)))  _Float16 v8h;
typedef __attribute__((ext_vector_type(2)))  __fp16   v2x;  // cvt_pkrtz result
typedef __attribute__((ext_vector_type(8)))  float    v8f;
typedef __attribute__((ext_vector_type(4)))  float    v4f;

union F16x16 { v16h v; v8h h[2]; _Float16 e[16]; };
union F16x8  { v8h  v; _Float16 e[8]; };
union F16x8p { v8h  v; v2x p[4]; };
union H2     { _Float16 e[2]; unsigned u; };

__device__ __forceinline__ v8f wmma_f16(v16h a, v16h b, v8f c) {
  // D = A(16x32 f16) * B(32x16 f16) + C(16x16 f32)
  return __builtin_amdgcn_wmma_f32_16x16x32_f16(
      /*neg_a=*/false, a, /*neg_b=*/false, b,
      /*c_mod=*/(short)0, c, /*reuse_a=*/false, /*reuse_b=*/false);
}

__device__ __forceinline__ v8f vzero8() {
  v8f z;
#pragma unroll
  for (int i = 0; i < 8; ++i) z[i] = 0.0f;
  return z;
}

// Async copy 16B global -> LDS (GLOBAL_LOAD_ASYNC_TO_LDS_B128, ASYNCcnt).
// Generic LDS pointer low 32 bits = LDS byte offset (flat LDS aperture rule).
__device__ __forceinline__ void async_copy_b128(const void* gptr, void* lptr) {
  unsigned lds = (unsigned)(unsigned long long)(uintptr_t)lptr;
  asm volatile("global_load_async_to_lds_b128 %0, %1, off"
               :: "v"(lds), "v"(gptr) : "memory");
}
__device__ __forceinline__ void async_wait0() {
  asm volatile("s_wait_asynccnt 0" ::: "memory");
}

// ---------------------------------------------------------------------------
// Tiled GEMM: C[M x N] = A[M x K] * B[K x N] (+ bias)
//   A: f16 (A_F16=true, async-staged) or f32 (packed-convert staged).
//   B: always f32 (weights), staged transposed [n][k] with packed k-pairs.
//   Out: f16 workspace (OUT_F32=false) or f32 + bias (OUT_F32=true).
// Block = 256 threads = 8 waves; block tile 128x128; wave tile 32x64.
// ---------------------------------------------------------------------------
template <bool A_F16, bool OUT_F32>
__global__ __launch_bounds__(256) void gemm_kernel(
    const void* __restrict__ Aptr, const float* __restrict__ Bptr,
    _Float16* __restrict__ Ch, float* __restrict__ Cf,
    const float* __restrict__ bias, int M, int Nn, int K) {
  __shared__ _Float16 ldsA[128 * 32];   // [m][k]
  __shared__ _Float16 ldsB[128 * 32];   // [n][k]  (transposed B chunk)

  const int tid  = threadIdx.x;
  const int wave = tid >> 5;
  const int lane = tid & 31;
  const int l16  = lane & 15;
  const int hi   = lane >> 4;           // lane half: 0 or 1
  const int mBlock = blockIdx.y * 128;
  const int nBlock = blockIdx.x * 128;
  const int wm = (wave >> 1) * 32;      // 4 wave-rows of 32
  const int wn = (wave & 1) * 64;       // 2 wave-cols of 64

  v8f acc[2][4];
#pragma unroll
  for (int tm = 0; tm < 2; ++tm)
#pragma unroll
    for (int tn = 0; tn < 4; ++tn) acc[tm][tn] = vzero8();

  for (int k0 = 0; k0 < K; k0 += 32) {
    __syncthreads();
    // ---- stage A chunk: 128 rows x 32 k ----------------------------------
    {
      const int row = tid >> 1;
      const int cb  = (tid & 1) * 16;
      if constexpr (A_F16) {
        const _Float16* Ag =
            (const _Float16*)Aptr + (size_t)(mBlock + row) * K + k0 + cb;
        async_copy_b128(Ag,     &ldsA[row * 32 + cb]);
        async_copy_b128(Ag + 8, &ldsA[row * 32 + cb + 8]);
      } else {
        const float* Ag =
            (const float*)Aptr + (size_t)(mBlock + row) * K + k0 + cb;
        v4f a0 = *(const v4f*)Ag,       a1 = *(const v4f*)(Ag + 4);
        v4f a2 = *(const v4f*)(Ag + 8), a3 = *(const v4f*)(Ag + 12);
        F16x8p t0, t1;
        t0.p[0] = __builtin_amdgcn_cvt_pkrtz(a0[0], a0[1]);
        t0.p[1] = __builtin_amdgcn_cvt_pkrtz(a0[2], a0[3]);
        t0.p[2] = __builtin_amdgcn_cvt_pkrtz(a1[0], a1[1]);
        t0.p[3] = __builtin_amdgcn_cvt_pkrtz(a1[2], a1[3]);
        t1.p[0] = __builtin_amdgcn_cvt_pkrtz(a2[0], a2[1]);
        t1.p[1] = __builtin_amdgcn_cvt_pkrtz(a2[2], a2[3]);
        t1.p[2] = __builtin_amdgcn_cvt_pkrtz(a3[0], a3[1]);
        t1.p[3] = __builtin_amdgcn_cvt_pkrtz(a3[2], a3[3]);
        *(v8h*)&ldsA[row * 32 + cb]     = t0.v;
        *(v8h*)&ldsA[row * 32 + cb + 8] = t1.v;
        if (k0 + 32 < K) __builtin_prefetch(Ag + 32, 0, 1);
      }
    }
    // ---- stage B transposed: ldsB[n][k], packed (k,k+1) pairs ------------
    {
      const int kp = (tid >> 4) * 2;      // 0,2,...,30
      const int nb = (tid & 15) * 8;      // 0..120
      const float* B0 = Bptr + (size_t)(k0 + kp) * Nn + nBlock + nb;
      const float* B1 = B0 + Nn;
      v4f x0 = *(const v4f*)B0, x1 = *(const v4f*)(B0 + 4);
      v4f y0 = *(const v4f*)B1, y1 = *(const v4f*)(B1 + 4);
#pragma unroll
      for (int i = 0; i < 4; ++i) {
        *(v2x*)&ldsB[(nb + i) * 32 + kp] =
            __builtin_amdgcn_cvt_pkrtz(x0[i], y0[i]);
        *(v2x*)&ldsB[(nb + 4 + i) * 32 + kp] =
            __builtin_amdgcn_cvt_pkrtz(x1[i], y1[i]);
      }
      if (k0 + 32 < K) __builtin_prefetch(B0 + (size_t)32 * Nn, 0, 1);
    }
    if constexpr (A_F16) async_wait0();
    __syncthreads();

    // ---- fragments (ISA VGPR layouts) ------------------------------------
    // A 16x32: lane=row(l16); elems 0..7 -> k = hi*8+0..7 ; 8..15 -> +16
    F16x16 af[2];
#pragma unroll
    for (int tm = 0; tm < 2; ++tm) {
      const _Float16* p = &ldsA[(wm + tm * 16 + l16) * 32 + hi * 8];
      af[tm].h[0] = *(const v8h*)p;
      af[tm].h[1] = *(const v8h*)(p + 16);
    }
    // B 32x16: lane=col(l16); elems 0..7 -> k = hi*8+0..7 ; 8..15 -> +16
    F16x16 bf[4];
#pragma unroll
    for (int tn = 0; tn < 4; ++tn) {
      const _Float16* p = &ldsB[(wn + tn * 16 + l16) * 32 + hi * 8];
      bf[tn].h[0] = *(const v8h*)p;
      bf[tn].h[1] = *(const v8h*)(p + 16);
    }
#pragma unroll
    for (int tm = 0; tm < 2; ++tm)
#pragma unroll
      for (int tn = 0; tn < 4; ++tn)
        acc[tm][tn] = wmma_f16(af[tm].v, bf[tn].v, acc[tm][tn]);
  }

  // ---- epilogue: C layout lane=col(l16), VGPR r -> row r + hi*8 ----------
#pragma unroll
  for (int tm = 0; tm < 2; ++tm)
#pragma unroll
    for (int tn = 0; tn < 4; ++tn) {
      const int col     = nBlock + wn + tn * 16 + l16;
      const int rowbase = mBlock + wm + tm * 16 + hi * 8;
      if constexpr (OUT_F32) {
        const float bv = bias[col];
#pragma unroll
        for (int r = 0; r < 8; ++r)
          Cf[(size_t)(rowbase + r) * Nn + col] = acc[tm][tn][r] + bv;
      } else {
#pragma unroll
        for (int r = 0; r < 8; ++r)
          Ch[(size_t)(rowbase + r) * Nn + col] = (_Float16)acc[tm][tn][r];
      }
    }
}

// ---------------------------------------------------------------------------
// Flash attention. grid = (8 q-tiles, 96 batch*head), block = 256 (8 waves).
// Each wave owns 16 Q rows; K/V processed in 128-row chunks from LDS.
// K tile staged with global_load_async_to_lds_b128 (pure f16 copy).
// ---------------------------------------------------------------------------
__global__ __launch_bounds__(256) void attn_kernel(
    const _Float16* __restrict__ qkv, _Float16* __restrict__ attn_out) {
  __shared__ _Float16 ldsK[128 * 64];        // [krow][d]
  __shared__ _Float16 ldsVt[64 * 128];       // [d][krow]  (transposed)
  __shared__ _Float16 ldsP[8 * 16 * 128];    // per-wave P strip [16][128]

  const int tid  = threadIdx.x;
  const int wave = tid >> 5;
  const int lane = tid & 31;
  const int l16  = lane & 15;
  const int hi   = lane >> 4;
  const int bh   = blockIdx.y;
  const int b    = bh / NHEADS;
  const int h    = bh % NHEADS;
  const int qt   = blockIdx.x;

  const _Float16* qbase = qkv + (size_t)(b * SEQ) * QKVD + h * HDIM;
  const _Float16* kbase = qbase + EMB;
  const _Float16* vbase = qbase + 2 * EMB;

  // Q fragments (A layout), scale 1/sqrt(64)=0.125 folded in (exact in f16).
  F16x16 aq[2];
  {
    const int qrow = qt * 128 + wave * 16 + l16;
#pragma unroll
    for (int c = 0; c < 2; ++c) {
      const _Float16* p = qbase + (size_t)qrow * QKVD + c * 32 + hi * 8;
      aq[c].h[0] = *(const v8h*)p;
      aq[c].h[1] = *(const v8h*)(p + 16);
#pragma unroll
      for (int i = 0; i < 16; ++i) aq[c].e[i] = aq[c].e[i] * (_Float16)0.125f;
    }
  }

  v8f o[4];
#pragma unroll
  for (int t = 0; t < 4; ++t) o[t] = vzero8();
  float mrow[8], lrow[8];
#pragma unroll
  for (int r = 0; r < 8; ++r) { mrow[r] = -1e30f; lrow[r] = 0.0f; }

  for (int kc = 0; kc < SEQ / 128; ++kc) {
    __syncthreads();
    // ---- stage K rows via async-to-LDS (straight f16 copy) ---------------
    {
      const int row = tid >> 1;
      const int cb  = (tid & 1) * 32;
      const _Float16* ks = kbase + (size_t)(kc * 128 + row) * QKVD + cb;
#pragma unroll
      for (int q = 0; q < 4; ++q)
        async_copy_b128(ks + q * 8, &ldsK[row * 64 + cb + q * 8]);
    }
    // ---- stage V transposed: packed (row,row+1) pairs, b32 stores --------
    {
      const int row2 = (tid >> 2) * 2;       // 0,2,...,126
      const int db   = (tid & 3) * 16;       // 0,16,32,48
      const _Float16* v0 = vbase + (size_t)(kc * 128 + row2) * QKVD + db;
      const _Float16* v1 = v0 + QKVD;
      F16x8 a0, a1, b0, b1;
      a0.v = *(const v8h*)v0; a1.v = *(const v8h*)(v0 + 8);
      b0.v = *(const v8h*)v1; b1.v = *(const v8h*)(v1 + 8);
#pragma unroll
      for (int i = 0; i < 8; ++i) {
        H2 p0; p0.e[0] = a0.e[i]; p0.e[1] = b0.e[i];
        *(unsigned*)&ldsVt[(db + i) * 128 + row2] = p0.u;
        H2 p1; p1.e[0] = a1.e[i]; p1.e[1] = b1.e[i];
        *(unsigned*)&ldsVt[(db + 8 + i) * 128 + row2] = p1.u;
      }
    }
    async_wait0();
    __syncthreads();

    // ---- S = Q * K^T : 8 col-tiles x 2 k-chunks --------------------------
    v8f s[8];
#pragma unroll
    for (int j = 0; j < 8; ++j) s[j] = vzero8();
#pragma unroll
    for (int j = 0; j < 8; ++j)
#pragma unroll
      for (int c = 0; c < 2; ++c) {
        F16x16 bk;
        const _Float16* p = &ldsK[(j * 16 + l16) * 64 + c * 32 + hi * 8];
        bk.h[0] = *(const v8h*)p;
        bk.h[1] = *(const v8h*)(p + 16);
        s[j] = wmma_f16(aq[c].v, bk.v, s[j]);
      }

    // ---- online softmax (rows r + hi*8, cols striped over 16 lanes) ------
    _Float16* pbase = &ldsP[(wave * 16) * 128];
#pragma unroll
    for (int r = 0; r < 8; ++r) {
      float mx = s[0][r];
#pragma unroll
      for (int j = 1; j < 8; ++j) mx = fmaxf(mx, s[j][r]);
#pragma unroll
      for (int msk = 1; msk < 16; msk <<= 1)
        mx = fmaxf(mx, __shfl_xor(mx, msk, 32));
      const float mnew  = fmaxf(mrow[r], mx);
      const float alpha = exp2f((mrow[r] - mnew) * L2E);
      float rsum = 0.0f;
      _Float16* prow = pbase + (r + hi * 8) * 128;
#pragma unroll
      for (int j = 0; j < 8; ++j) {
        const float pv = exp2f((s[j][r] - mnew) * L2E);
        rsum += pv;
        prow[j * 16 + l16] = (_Float16)pv;
      }
#pragma unroll
      for (int msk = 1; msk < 16; msk <<= 1)
        rsum += __shfl_xor(rsum, msk, 32);
      lrow[r] = lrow[r] * alpha + rsum;
      mrow[r] = mnew;
#pragma unroll
      for (int t = 0; t < 4; ++t) o[t][r] *= alpha;
    }

    // ---- O += P * V : 4 out col-tiles x 4 k-subchunks --------------------
#pragma unroll
    for (int t = 0; t < 4; ++t)
#pragma unroll
      for (int sub = 0; sub < 4; ++sub) {
        F16x16 ap, bv;
        const _Float16* pp = pbase + l16 * 128 + sub * 32 + hi * 8;
        ap.h[0] = *(const v8h*)pp;
        ap.h[1] = *(const v8h*)(pp + 16);
        const _Float16* vp = &ldsVt[(t * 16 + l16) * 128 + sub * 32 + hi * 8];
        bv.h[0] = *(const v8h*)vp;
        bv.h[1] = *(const v8h*)(vp + 16);
        o[t] = wmma_f16(ap.v, bv.v, o[t]);
      }
  }

  // ---- normalize and store f16 -------------------------------------------
#pragma unroll
  for (int r = 0; r < 8; ++r) {
    const float inv = 1.0f / lrow[r];
    const int row = qt * 128 + wave * 16 + r + hi * 8;
    _Float16* dst = attn_out + (size_t)(b * SEQ + row) * EMB + h * HDIM;
#pragma unroll
    for (int t = 0; t < 4; ++t)
      dst[t * 16 + l16] = (_Float16)(o[t][r] * inv);
  }
}

// ---------------------------------------------------------------------------
extern "C" void kernel_launch(void* const* d_in, const int* in_sizes, int n_in,
                              void* d_out, int out_size, void* d_ws,
                              size_t ws_size, hipStream_t stream) {
  const float* x     = (const float*)d_in[0];
  const float* w_qkv = (const float*)d_in[1];
  const float* w_out = (const float*)d_in[2];
  const float* b_out = (const float*)d_in[3];
  float* out = (float*)d_out;

  _Float16* qkv_h  = (_Float16*)d_ws;                                // 37.75 MB
  _Float16* attn_h = (_Float16*)((char*)d_ws +
                                 (size_t)ROWS * QKVD * sizeof(_Float16));

  // 1) QKV projection: [8192x768] * [768x2304] -> f16
  gemm_kernel<false, false><<<dim3(QKVD / 128, ROWS / 128), 256, 0, stream>>>(
      (const void*)x, w_qkv, qkv_h, nullptr, nullptr, ROWS, QKVD, EMB);

  // 2) Attention per (batch, head): 8 q-tiles x 96 (b,h)
  attn_kernel<<<dim3(SEQ / 128, BATCH * NHEADS), 256, 0, stream>>>(qkv_h,
                                                                   attn_h);

  // 3) Output projection + bias: [8192x768] * [768x768] -> f32
  gemm_kernel<true, true><<<dim3(EMB / 128, ROWS / 128), 256, 0, stream>>>(
      (const void*)attn_h, w_out, nullptr, out, b_out, ROWS, EMB, EMB);
}
